// Net_19628000543058
// MI455X (gfx1250) — matrix operand order repaired and checked by reading, not compiled, not run on hardware
//
#include <hip/hip_runtime.h>
#include <cstdint>
#include <cstddef>

// ---------- types ----------
typedef __attribute__((ext_vector_type(8)))  __bf16 v8bf;
typedef __attribute__((ext_vector_type(16))) __bf16 v16bf;
typedef __attribute__((ext_vector_type(8)))  float  v8f;
typedef __attribute__((ext_vector_type(4)))  unsigned int u32x4;
typedef __attribute__((ext_vector_type(8)))  int i32x8;
typedef __attribute__((ext_vector_type(4)))  int i32x4;

#define BATCH  512
#define TILE_K 32

#if defined(__has_builtin)
#if __has_builtin(__builtin_amdgcn_tensor_load_to_lds) && __has_builtin(__builtin_amdgcn_s_wait_tensorcnt)
#define HAVE_TDM 1
#endif
#endif

__device__ __forceinline__ unsigned short f2bf(float f) {
  // round-to-nearest-even fp32 -> bf16 (bit trick, no hw cvt dependence)
  unsigned int u = __builtin_bit_cast(unsigned int, f);
  unsigned int r = u + 0x7FFFu + ((u >> 16) & 1u);
  return (unsigned short)(r >> 16);
}
__device__ __forceinline__ float bf2f(unsigned short s) {
  unsigned int u = ((unsigned int)s) << 16;
  return __builtin_bit_cast(float, u);
}

#ifdef HAVE_TDM
// Issue a TDM 2D tile load: tile (TILE_K x BATCH) of 2-byte elements from a
// (Kpad x BATCH) row-major bf16 tensor into LDS at ldsOff. Wave-level op.
// clang-23 toolchain: 6-arg builtin (g0, g1, g2, g3, g4[VADDR4=null], cpol).
__device__ __forceinline__ void tdm_load_tile(const unsigned short* gbase, int Kpad,
                                              int k0, unsigned ldsOff)
{
  unsigned long long ga =
      (unsigned long long)(uintptr_t)gbase + (unsigned long long)k0 * 2ull;
  u32x4 g0;
  g0.x = 1u;                                   // count=1, user descriptor
  g0.y = ldsOff;                               // LDS byte address
  g0.z = (unsigned)(ga & 0xFFFFFFFFu);         // global_addr[31:0]
  g0.w = (unsigned)((ga >> 32) & 0x01FFFFFFu)  // global_addr[56:32]
         | 0x80000000u;                        // type=2 ("image") in [127:126]
  i32x8 g1 = {
    0x00010000,                                      // data_size=1 (2 bytes)
    (int)(((unsigned)Kpad & 0xFFFFu) << 16),         // tensor_dim0[15:0]
    (int)((((unsigned)Kpad >> 16) & 0xFFFFu)
          | ((unsigned)BATCH << 16)),                // d0[31:16] | d1[15:0]
    (int)((TILE_K) << 16),                           // d1[31:16]=0 | tile_dim0
    (int)(BATCH),                                    // tile_dim1=512, tile_dim2=0
    (int)(Kpad),                                     // tensor_dim0_stride[31:0]
    0,                                               // stride0[47:32] | stride1 lo
    0
  };
  i32x4 z4 = {0, 0, 0, 0};
  i32x8 z8 = {0, 0, 0, 0, 0, 0, 0, 0};
  __builtin_amdgcn_tensor_load_to_lds(g0, g1, z4, z4, z8, 0);
}
#endif

// ----------------------------------------------------------------------------
// GEMM: Z(512 x O) = A(512 x K, bf16, row stride Kpad) @ W(O x K, f32)^T + bias
// One WG = 256 threads = 8 waves. WG tile: M=512, N=64. Wave tile: 64x64
// (4x4 WMMA tiles of 16x16, K-step 32, v_wmma_f32_16x16x32_bf16).
// Double-buffered software pipeline: TDM DMA of A tile ks+1 + weight-stream
// conversion for ks+1 overlap the 16 WMMAs of step ks; one barrier per step.
// ----------------------------------------------------------------------------
__global__ __launch_bounds__(256)
void gemm_wmma(const unsigned short* __restrict__ A, int K, int Kpad,
               const float* __restrict__ W, const float* __restrict__ bias,
               float* __restrict__ Z, int O)
{
  __shared__ __align__(16) unsigned short As[2][BATCH * TILE_K]; // 64 KB
  __shared__ __align__(16) unsigned short Bs[2][64 * TILE_K];    //  8 KB

  const int tid  = threadIdx.x;
  const int lane = tid & 31;
  const int wave = tid >> 5;        // 0..7 -> rows 64*wave .. 64*wave+63
  const int lh   = lane >> 4;       // half-wave select
  const int l15  = lane & 15;
  const int n0   = blockIdx.x * 64;

#ifdef HAVE_TDM
  const unsigned asOff0 = (unsigned)(uintptr_t)(&As[0][0]); // flat->LDS offset
  const unsigned asOff1 = (unsigned)(uintptr_t)(&As[1][0]);
#endif

  // ---- weight tile loader: 64 rows of W, 32 k, fp32 -> bf16 into Bs[buf] ----
  auto loadB = [&](int ksArg, int buf) {
    const int k0b = ksArg * TILE_K;
    const bool tail = (k0b + TILE_K > K);
    #pragma unroll
    for (int c = 0; c < 2; ++c) {
      int chunk = tid + c * 256;              // 0..511
      int r  = chunk >> 3;
      int kc = (chunk & 7) * 4;
      int o  = n0 + r;
      float f0 = 0.f, f1 = 0.f, f2 = 0.f, f3 = 0.f;
      if (o < O) {
        const float* wp = W + (long long)o * K + k0b + kc;
        if (!tail) {
          f0 = wp[0]; f1 = wp[1]; f2 = wp[2]; f3 = wp[3];
          if (k0b + 2 * TILE_K <= K) __builtin_prefetch(wp + TILE_K, 0, 3);
        } else {
          int kb = k0b + kc;
          if (kb + 0 < K) f0 = wp[0];
          if (kb + 1 < K) f1 = wp[1];
          if (kb + 2 < K) f2 = wp[2];
          if (kb + 3 < K) f3 = wp[3];
        }
      }
      uint2 pk;
      pk.x = (unsigned int)f2bf(f0) | ((unsigned int)f2bf(f1) << 16);
      pk.y = (unsigned int)f2bf(f2) | ((unsigned int)f2bf(f3) << 16);
      *(uint2*)(&Bs[buf][r * TILE_K + kc]) = pk;
    }
  };

#ifndef HAVE_TDM
  // fallback A tile loader: 512x32 bf16, raw copy global->LDS
  auto loadA = [&](int ksArg, int buf) {
    const int k0a = ksArg * TILE_K;
    #pragma unroll
    for (int c = 0; c < 8; ++c) {
      int chunk = tid + c * 256;              // 0..2047
      int row = chunk >> 2;
      int kc  = (chunk & 3) * 8;
      *(uint4*)(&As[buf][row * TILE_K + kc]) =
          *(const uint4*)(&A[(size_t)row * Kpad + k0a + kc]);
    }
  };
#endif

  const v8f vzero = {0.f,0.f,0.f,0.f,0.f,0.f,0.f,0.f};
  v8f acc[4][4];
  #pragma unroll
  for (int i = 0; i < 4; ++i)
    #pragma unroll
    for (int j = 0; j < 4; ++j) acc[i][j] = vzero;

  const int ksteps = Kpad / TILE_K;

  // ---- prologue: fill buffer 0 ----
#ifdef HAVE_TDM
  if (tid < 32) tdm_load_tile(A, Kpad, 0, asOff0);
#else
  loadA(0, 0);
#endif
  loadB(0, 0);
#ifdef HAVE_TDM
  if (tid < 32) __builtin_amdgcn_s_wait_tensorcnt(0);
#endif
  __syncthreads();

  for (int ks = 0; ks < ksteps; ++ks) {
    const int cur = ks & 1;
    const int nxt = cur ^ 1;

    // ---- stage ks+1 into the alternate buffers (overlaps compute) ----
    if (ks + 1 < ksteps) {
#ifdef HAVE_TDM
      if (tid < 32)
        tdm_load_tile(A, Kpad, (ks + 1) * TILE_K, nxt ? asOff1 : asOff0);
#else
      loadA(ks + 1, nxt);
#endif
      loadB(ks + 1, nxt);
    }

    // ---- B fragments (32x16 bf16): lane<16 -> N=lane, K {0..7,16..23} ;
    //      lane>=16 -> N=lane-16, K {8..15,24..31} ----
    v16bf bfrag[4];
    #pragma unroll
    for (int nj = 0; nj < 4; ++nj) {
      const v8bf* q = (const v8bf*)(const void*)
          (&Bs[cur][(nj * 16 + l15) * TILE_K + lh * 8]);
      v8bf lo = q[0], hi = q[2];
      bfrag[nj] = __builtin_shufflevector(lo, hi,
          0,1,2,3,4,5,6,7,8,9,10,11,12,13,14,15);
    }

    // ---- A fragments (16x32 bf16): lane<16 -> K {0..7,16..23},
    //      lane>=16 -> K {8..15,24..31}; M = lane&15 ----
    #pragma unroll
    for (int mi = 0; mi < 4; ++mi) {
      int row = wave * 64 + mi * 16 + l15;
      const v8bf* p0 = (const v8bf*)(const void*)
          (&As[cur][row * TILE_K + lh * 8]);
      const v8bf* p1 = (const v8bf*)(const void*)
          (&As[cur][row * TILE_K + lh * 8 + 16]);
      v8bf lo = *p0, hi = *p1;
      v16bf afrag = __builtin_shufflevector(lo, hi,
          0,1,2,3,4,5,6,7,8,9,10,11,12,13,14,15);
      #pragma unroll
      for (int nj = 0; nj < 4; ++nj) {
        acc[mi][nj] = __builtin_amdgcn_wmma_f32_16x16x32_bf16(
            false, afrag, false, bfrag[nj],
            (short)0, acc[mi][nj], false, false);
      }
    }

#ifdef HAVE_TDM
    if (tid < 32) __builtin_amdgcn_s_wait_tensorcnt(0); // tile ks+1 landed
#endif
    __syncthreads();  // publish ks+1 buffers; everyone done reading cur
  }

  // ---- store D + bias: VGPR r -> M = r + 8*lh ; N = l15 ----
  #pragma unroll
  for (int nj = 0; nj < 4; ++nj) {
    int col = n0 + nj * 16 + l15;
    if (col < O) {
      float bv = bias[col];
      #pragma unroll
      for (int mi = 0; mi < 4; ++mi) {
        int rb = wave * 64 + mi * 16 + lh * 8;
        #pragma unroll
        for (int r = 0; r < 8; ++r) {
          Z[(size_t)(rb + r) * O + col] = acc[mi][nj][r] + bv;
        }
      }
    }
  }
}

// ----------------------------------------------------------------------------
// BatchNorm (over batch dim) + ReLU, fp32 in -> padded bf16 out
// ----------------------------------------------------------------------------
__global__ void bn_relu(const float* __restrict__ Z, const float* __restrict__ g,
                        const float* __restrict__ be,
                        unsigned short* __restrict__ Hout, int O, int Kpad)
{
  int o = blockIdx.x * blockDim.x + threadIdx.x;
  if (o >= Kpad) return;
  if (o >= O) { // zero K-padding so GEMM needs no K guards on A
    for (int b = 0; b < BATCH; ++b) Hout[(size_t)b * Kpad + o] = 0;
    return;
  }
  float s = 0.f, s2 = 0.f;
  for (int b = 0; b < BATCH; ++b) {
    float z = Z[(size_t)b * O + o];
    s += z; s2 += z * z;
  }
  float m   = s  * (1.0f / BATCH);
  float var = s2 * (1.0f / BATCH) - m * m;
  float inv = rsqrtf(var + 1e-5f);
  float gg  = g[o] * inv;
  float bo  = be[o] - m * gg;
  for (int b = 0; b < BATCH; ++b) {
    float z = Z[(size_t)b * O + o] * gg + bo;
    Hout[(size_t)b * Kpad + o] = f2bf(fmaxf(z, 0.f));
  }
}

// fp32 input -> padded bf16
__global__ void cvt_pad(const float* __restrict__ X,
                        unsigned short* __restrict__ Xb, int K, int Kpad)
{
  int i = blockIdx.x * blockDim.x + threadIdx.x;
  if (i >= BATCH * Kpad) return;
  int row = i / Kpad, k = i - row * Kpad;
  Xb[i] = (k < K) ? f2bf(X[row * K + k]) : (unsigned short)0;
}

// ----------------------------------------------------------------------------
// head: logits = H7(512x128 bf16) @ Wd(Dx128)^T + bd ; row softmax -> P(512xD)
// ----------------------------------------------------------------------------
__global__ void head_softmax(const unsigned short* __restrict__ H7,
                             const float* __restrict__ Wd,
                             const float* __restrict__ bd,
                             float* __restrict__ P, int D)
{
  int b = blockIdx.x * blockDim.x + threadIdx.x;
  if (b >= BATCH) return;
  float acc[8];
  #pragma unroll
  for (int d = 0; d < 8; ++d) acc[d] = (d < D) ? bd[d] : -3.0e38f;
  for (int k = 0; k < 128; ++k) {
    float h = bf2f(H7[(size_t)b * 128 + k]);
    #pragma unroll
    for (int d = 0; d < 8; ++d)
      if (d < D) acc[d] = fmaf(h, Wd[d * 128 + k], acc[d]);
  }
  float mx = -3.0e38f;
  #pragma unroll
  for (int d = 0; d < 8; ++d) if (d < D) mx = fmaxf(mx, acc[d]);
  float sum = 0.f;
  #pragma unroll
  for (int d = 0; d < 8; ++d)
    if (d < D) { acc[d] = expf(acc[d] - mx); sum += acc[d]; }
  float inv = 1.0f / sum;
  #pragma unroll
  for (int d = 0; d < 8; ++d) if (d < D) P[b * D + d] = acc[d] * inv;
}

// ----------------------------------------------------------------------------
// power-iteration whitening: single wave32 block (serial by construction)
// ----------------------------------------------------------------------------
__global__ void power_whiten(const float* __restrict__ P,
                             const float* __restrict__ R,
                             float* __restrict__ out, int D)
{
  __shared__ float C[64];
  __shared__ float Wm[64];
  __shared__ float v[8];
  __shared__ float u[8];
  int t = threadIdx.x;

  // C = P^T P / B
  for (int p = t; p < D * D; p += 32) {
    int i = p / D, j = p % D;
    float s = 0.f;
    for (int b = 0; b < BATCH; ++b) s += P[b * D + i] * P[b * D + j];
    C[p] = s * (1.0f / BATCH);
    Wm[p] = 0.f;
  }
  __syncthreads();

  for (int it = 0; it < D; ++it) {
    if (t < D) v[t] = R[t * D + it];   // v0 = R[:, it]
    __syncthreads();
    for (int n = 0; n < 250; ++n) {
      float uu = 0.f;
      if (t < D) for (int k = 0; k < D; ++k) uu += C[t * D + k] * v[k];
      __syncthreads();
      if (t < D) u[t] = uu;
      __syncthreads();
      float nr = 0.f;
      for (int k = 0; k < D; ++k) nr += u[k] * u[k];
      nr = sqrtf(nr);
      if (t < D) v[t] = uu / nr;
      __syncthreads();
    }
    // l = ||C v|| ; deflate ; accumulate W
    float uu = 0.f;
    if (t < D) for (int k = 0; k < D; ++k) uu += C[t * D + k] * v[k];
    __syncthreads();
    if (t < D) u[t] = uu;
    __syncthreads();
    float l = 0.f;
    for (int k = 0; k < D; ++k) l += u[k] * u[k];
    l = sqrtf(l);
    float isl = 1.0f / sqrtf(l);
    for (int p = t; p < D * D; p += 32) {
      int i = p / D, j = p % D;
      float vv = v[i] * v[j];
      C[p]  -= l * vv;
      Wm[p] += vv * isl;
    }
    __syncthreads();
  }
  // out = P @ W^T
  for (int b = t; b < BATCH; b += 32) {
    for (int i = 0; i < D; ++i) {
      float s = 0.f;
      for (int j = 0; j < D; ++j) s += P[b * D + j] * Wm[i * D + j];
      out[b * D + i] = s;
    }
  }
}

// ----------------------------------------------------------------------------
extern "C" void kernel_launch(void* const* d_in, const int* in_sizes, int n_in,
                              void* d_out, int out_size, void* d_ws, size_t ws_size,
                              hipStream_t stream)
{
  (void)in_sizes; (void)n_in; (void)out_size; (void)ws_size;
  const float *w[7], *bb[7], *g[7], *be[7];
  for (int li = 0; li < 7; ++li) {
    w[li]  = (const float*)d_in[li * 4 + 0];
    bb[li] = (const float*)d_in[li * 4 + 1];
    g[li]  = (const float*)d_in[li * 4 + 2];
    be[li] = (const float*)d_in[li * 4 + 3];
  }
  const float* w8 = (const float*)d_in[28];
  const float* b8 = (const float*)d_in[29];
  const float* w9 = (const float*)d_in[30];
  const float* b9 = (const float*)d_in[31];
  const float* R8 = (const float*)d_in[32];
  const float* R5 = (const float*)d_in[33];
  const float* x  = (const float*)d_in[34];

  const int H = 9000, DIN = 162;
  const int KpadX = 192, KpadH = 9024;   // multiples of 32

  char* wsb = (char*)d_ws;
  size_t bufBytes = (size_t)BATCH * KpadH * sizeof(unsigned short); // 9.24 MB
  unsigned short* hbuf0 = (unsigned short*)wsb;
  unsigned short* hbuf1 = (unsigned short*)(wsb + bufBytes);
  float* Z  = (float*)(wsb + 2 * bufBytes);                          // 18.4 MB
  float* P8 = (float*)(wsb + 2 * bufBytes + (size_t)BATCH * H * sizeof(float));
  float* P5 = P8 + BATCH * 8;

  dim3 blk(256);
  // input -> padded bf16
  cvt_pad<<<dim3((BATCH * KpadX + 255) / 256), blk, 0, stream>>>(x, hbuf0, DIN, KpadX);

  // layer 1: 162 -> 9000
  gemm_wmma<<<dim3((H + 63) / 64), blk, 0, stream>>>(hbuf0, DIN, KpadX, w[0], bb[0], Z, H);
  bn_relu<<<dim3((KpadH + 255) / 256), blk, 0, stream>>>(Z, g[0], be[0], hbuf1, H, KpadH);

  // layers 2..6: 9000 -> 9000 (ping-pong)
  unsigned short* hin = hbuf1; unsigned short* hout = hbuf0;
  for (int li = 1; li < 6; ++li) {
    gemm_wmma<<<dim3((H + 63) / 64), blk, 0, stream>>>(hin, H, KpadH, w[li], bb[li], Z, H);
    bn_relu<<<dim3((KpadH + 255) / 256), blk, 0, stream>>>(Z, g[li], be[li], hout, H, KpadH);
    unsigned short* tmp = hin; hin = hout; hout = tmp;
  }

  // layer 7: 9000 -> 128
  gemm_wmma<<<dim3((128 + 63) / 64), blk, 0, stream>>>(hin, H, KpadH, w[6], bb[6], Z, 128);
  bn_relu<<<dim3(1), blk, 0, stream>>>(Z, g[6], be[6], hout, 128, 128);

  // heads: softmax projections
  head_softmax<<<dim3(2), blk, 0, stream>>>(hout, w8, b8, P8, 8);
  head_softmax<<<dim3(2), blk, 0, stream>>>(hout, w9, b9, P5, 5);

  // power whitening -> outputs (x1 then x2, flat)
  float* out = (float*)d_out;
  power_whiten<<<dim3(1), dim3(32), 0, stream>>>(P8, R8, out, 8);
  power_whiten<<<dim3(1), dim3(32), 0, stream>>>(P5, R5, out + BATCH * 8, 5);
}
